// MoELinearRegression_11029476016646
// MI455X (gfx1250) — matrix operand order repaired and checked by reading, not compiled, not run on hardware
//
#include <hip/hip_runtime.h>
#include <hip/hip_bf16.h>
#include <math.h>

// ---------------------------------------------------------------------------
// MoE linear regression, fused for MI455X (gfx1250, wave32).
//
//   nl = x @ Wn.T + bn ; eo = x @ We.T ; out = sum(top8-softmax(nl) * eo)
//
// GEMMs run on the bf16 WMMA pipe with an f32-accurate hi/lo split:
//   x = xh + xl, W = wh + wl (bf16),  x.W ~= xh.wh + xh.wl + xl.wh
// 3x bf16 16x16x32 WMMAs replace 8x f32 16x16x4 WMMAs -> ~2.7x matrix-pipe
// cycles saved; kernel becomes HBM-bound (268 MB of x @ 23.3 TB/s ~ 11.5us).
//
// CDNA5 paths: GLOBAL_LOAD_ASYNC_TO_LDS_B128 staging (ASYNCcnt) with
// prefetch-before-compute; weights pre-split to bf16 hi/lo in d_ws (L2).
// ---------------------------------------------------------------------------

typedef __attribute__((ext_vector_type(8)))  float  v8f;
typedef __attribute__((ext_vector_type(16))) __bf16 v16bf;
typedef __attribute__((ext_vector_type(8)))  __bf16 v8bf;
typedef __attribute__((ext_vector_type(4)))  __bf16 v4bf;
typedef __attribute__((__vector_size__(4 * sizeof(int)))) int v4i;

#define THREADS      256
#define ROWS_PER_BLK 128
#define D_DIM        1024
#define E_DIM        64
#define KC           64            // K chunk
#define XST          68            // f32 x-stage row stride (floats)
#define BST          72            // bf16 row stride (36 dwords, conflict-free)
// LDS map (float offsets):
#define XSTAGE_OFF   0             // 128*68            = 8704 f
#define XH_OFF       8704          // 128*72 bf16       = 4608 f
#define XL_OFF       13312         //                    = 4608 f
#define WB_OFF       17920         // 2 bufs * 4 flav * 64*72 bf16 = 2*9216 f
#define WBUF_BF      18432         // bf16 elems per weight buffer
#define WFLAV_BF     4608          // bf16 elems per flavor
#define SMEM_FLOATS  36352         // 145,408 bytes
#define NL_OFF       0             // phase-B overlay: 128*65
#define EO_OFF       8320
#define PB_STRIDE    65

#define PTR_G(p)   ((v4i*)(void*)(p))
#define PTR_LDS(p) ((__attribute__((address_space(3))) v4i*)(p))

// ---- prep: split Wn/We (f32) into bf16 hi/lo quadruple in d_ws -------------
__global__ __launch_bounds__(256)
void prep_weights(const float* __restrict__ Wn, const float* __restrict__ We,
                  __bf16* __restrict__ wsb)
{
    int idx = blockIdx.x * 256 + threadIdx.x;     // 0..65535
    float fn = Wn[idx], fe = We[idx];
    __bf16 nh = (__bf16)fn, eh = (__bf16)fe;
    wsb[idx]            = nh;
    wsb[idx +  65536]   = (__bf16)(fn - (float)nh);
    wsb[idx + 131072]   = eh;
    wsb[idx + 196608]   = (__bf16)(fe - (float)eh);
}

__global__ __launch_bounds__(THREADS)
void moe_fused_kernel(const float* __restrict__ x,
                      const float* __restrict__ bn,
                      const __bf16* __restrict__ wsb,
                      float* __restrict__ out)
{
    extern __shared__ float smem[];
    float*  xstage = smem + XSTAGE_OFF;
    __bf16* xh     = (__bf16*)(smem + XH_OFF);
    __bf16* xl     = (__bf16*)(smem + XL_OFF);
    __bf16* wb     = (__bf16*)(smem + WB_OFF);

    const int tid  = threadIdx.x;
    const int lane = tid & 31;
    const int w    = tid >> 5;
    const int l16  = lane & 15;
    const int hi   = lane >> 4;
    const int mg   = w >> 1;           // m-group 0..3  (32 rows each)
    const int ng   = w & 1;            // n-half  0..1  (32 experts each)
    const size_t row0 = (size_t)blockIdx.x * ROWS_PER_BLK;

    auto stage_x = [&](int kb) {       // x f32 chunk -> xstage (8 async/thr)
#pragma unroll
        for (int i = 0; i < 8; ++i) {
            int idx = tid + i * THREADS;
            int r = idx >> 4, c = (idx & 15) << 2;
            __builtin_amdgcn_global_load_async_to_lds_b128(
                PTR_G(&x[(row0 + r) * D_DIM + kb + c]),
                PTR_LDS(&xstage[r * XST + c]), 0, 0);
        }
    };
    auto stage_w = [&](int buf, int kb) {  // 4 bf16 flavors (8 async/thr)
        __bf16* dstb = wb + buf * WBUF_BF;
#pragma unroll
        for (int f = 0; f < 4; ++f) {
            const __bf16* srcb = wsb + f * 65536;
#pragma unroll
            for (int i = 0; i < 2; ++i) {
                int idx = tid + i * THREADS;         // 512 b128 per flavor
                int r = idx >> 3, c8 = (idx & 7) << 3;
                __builtin_amdgcn_global_load_async_to_lds_b128(
                    PTR_G(srcb + r * D_DIM + kb + c8),
                    PTR_LDS(dstb + f * WFLAV_BF + r * BST + c8), 0, 0);
            }
        }
    };
    auto convert_x = [&]() {           // xstage(f32) -> xh/xl (bf16)
#pragma unroll
        for (int i = 0; i < 8; ++i) {
            int idx = tid + i * THREADS;
            int r = idx >> 4, c = (idx & 15) << 2;
            float4 f = *(const float4*)&xstage[r * XST + c];
            v4bf h, l;
            h[0] = (__bf16)f.x; l[0] = (__bf16)(f.x - (float)h[0]);
            h[1] = (__bf16)f.y; l[1] = (__bf16)(f.y - (float)h[1]);
            h[2] = (__bf16)f.z; l[2] = (__bf16)(f.z - (float)h[2]);
            h[3] = (__bf16)f.w; l[3] = (__bf16)(f.w - (float)h[3]);
            *(v4bf*)(xh + r * BST + c) = h;
            *(v4bf*)(xl + r * BST + c) = l;
        }
    };

    // fragment loads: A = two 16B segs (K {0-7,16-23}/{8-15,24-31});
    //                 B = same two 16B segs (K {0-7,16-23}/{8-15,24-31})
    auto loadA = [](const __bf16* p) -> v16bf {
        v8bf s0 = *(const v8bf*)(p);
        v8bf s1 = *(const v8bf*)(p + 16);
        return __builtin_shufflevector(s0, s1, 0,1,2,3,4,5,6,7,
                                               8,9,10,11,12,13,14,15);
    };
    auto loadB = [](const __bf16* p) -> v16bf {
        v8bf s0 = *(const v8bf*)(p);
        v8bf s1 = *(const v8bf*)(p + 16);
        return __builtin_shufflevector(s0, s1, 0,1,2,3,4,5,6,7,
                                               8,9,10,11,12,13,14,15);
    };

    v8f accn[2][2], acce[2][2];
#pragma unroll
    for (int a = 0; a < 2; ++a)
#pragma unroll
        for (int b = 0; b < 2; ++b) { accn[a][b] = (v8f)0.0f; acce[a][b] = (v8f)0.0f; }

    auto compute_chunk = [&](const __bf16* wbase) {
#pragma unroll
        for (int ks = 0; ks < KC; ks += 32) {
            v16bf ah[2], al[2];
#pragma unroll
            for (int mt = 0; mt < 2; ++mt) {
                int ro = (mg * 32 + mt * 16 + l16) * BST + ks + 8 * hi;
                ah[mt] = loadA(xh + ro);
                al[mt] = loadA(xl + ro);
            }
#pragma unroll
            for (int mat = 0; mat < 2; ++mat) {
#pragma unroll
                for (int nt = 0; nt < 2; ++nt) {
                    int ro = (ng * 32 + nt * 16 + l16) * BST + ks + 8 * hi;
                    v16bf bh = loadB(wbase + (mat * 2 + 0) * WFLAV_BF + ro);
                    v16bf bl = loadB(wbase + (mat * 2 + 1) * WFLAV_BF + ro);
#pragma unroll
                    for (int mt = 0; mt < 2; ++mt) {
                        v8f* acc = mat ? &acce[mt][nt] : &accn[mt][nt];
                        *acc = __builtin_amdgcn_wmma_f32_16x16x32_bf16(
                            false, ah[mt], false, bh, (short)0, *acc, false, false);
                        *acc = __builtin_amdgcn_wmma_f32_16x16x32_bf16(
                            false, ah[mt], false, bl, (short)0, *acc, false, false);
                        *acc = __builtin_amdgcn_wmma_f32_16x16x32_bf16(
                            false, al[mt], false, bh, (short)0, *acc, false, false);
                    }
                }
            }
        }
    };

    // ---- pipeline: prefetch chunk k+1 before computing chunk k -------------
    stage_w(0, 0);
    stage_x(0);
    for (int kc = 0; kc < 16; ++kc) {
        __builtin_amdgcn_s_wait_asynccnt(0);   // chunk kc resident (prefetched)
        __syncthreads();
        convert_x();                           // xstage -> xh/xl
        __syncthreads();
        if (kc + 1 < 16) {                     // overlap with compute below
            stage_x((kc + 1) * KC);
            stage_w((kc + 1) & 1, (kc + 1) * KC);
        }
        compute_chunk(wb + (kc & 1) * WBUF_BF);
    }
    __syncthreads();                           // all computes done

    // ---- bias + scatter C fragments (rows m, cols n) -----------------------
    float bnv[2];
#pragma unroll
    for (int nt = 0; nt < 2; ++nt) bnv[nt] = bn[ng * 32 + nt * 16 + l16];

#pragma unroll
    for (int mt = 0; mt < 2; ++mt)
#pragma unroll
        for (int nt = 0; nt < 2; ++nt)
#pragma unroll
            for (int r = 0; r < 8; ++r) {
                int m = mg * 32 + mt * 16 + r + hi * 8;
                int n = ng * 32 + nt * 16 + l16;
                smem[NL_OFF + m * PB_STRIDE + n] = accn[mt][nt][r] + bnv[nt];
                smem[EO_OFF + m * PB_STRIDE + n] = acce[mt][nt][r];
            }
    __syncthreads();

    // ---- per-row top-8 + softmax + weighted sum ----------------------------
    if (tid < ROWS_PER_BLK) {
        const float* nlrow = &smem[NL_OFF + tid * PB_STRIDE];
        const float* eorow = &smem[EO_OFF + tid * PB_STRIDE];
        unsigned long long taken = 0ull;
        float m0 = 0.0f, denom = 0.0f, num = 0.0f;
        for (int p = 0; p < 8; ++p) {
            float mv = -3.4e38f;
            int   mi = 0;
            for (int j = 0; j < E_DIM; ++j) {
                float vv = nlrow[j];
                bool  sk = (taken >> j) & 1ull;
                if (!sk && vv > mv) { mv = vv; mi = j; }
            }
            taken |= (1ull << mi);
            if (p == 0) m0 = mv;
            float wgt = __expf(mv - m0);
            denom += wgt;
            num   += wgt * eorow[mi];
        }
        out[row0 + tid] = num / denom;
    }
}

extern "C" void kernel_launch(void* const* d_in, const int* in_sizes, int n_in,
                              void* d_out, int out_size, void* d_ws, size_t ws_size,
                              hipStream_t stream)
{
    // setup_inputs order: x, Wg, bg, Wn, bn, We, noise
    const float* x  = (const float*)d_in[0];
    const float* Wn = (const float*)d_in[3];
    const float* bn = (const float*)d_in[4];
    const float* We = (const float*)d_in[5];
    float* out = (float*)d_out;
    __bf16* wsb = (__bf16*)d_ws;     // 4 * 65536 bf16 = 512 KB scratch

    prep_weights<<<dim3(256), dim3(256), 0, stream>>>(Wn, We, wsb);

    const int B = 65536;
    size_t smem_bytes = SMEM_FLOATS * sizeof(float);   // 145,408 B
    moe_fused_kernel<<<dim3(B / ROWS_PER_BLK), dim3(THREADS), smem_bytes, stream>>>(
        x, bn, wsb, out);
}
